// LSTMUnroll_352187318643
// MI455X (gfx1250) — matrix operand order repaired and checked by reading, not compiled, not run on hardware
//
#include <hip/hip_runtime.h>

// ---------------------------------------------------------------------------
// LSTM unroll for MI455X (gfx1250): bf16 WMMA GEMM fused with LSTM pointwise.
// Per (t,layer) step, each block async-copies its input working set
// (A tile + 4 gate weight slabs, contiguous in global) into LDS via
// global_load_async_to_lds_b128 (ASYNCcnt), then runs the WMMA loop entirely
// out of LDS. Weights stay L2-resident (40MB bf16 < 192MB L2). Layer 0
// converts x_t fp32->bf16 inline during staging (no separate conv kernel).
// ---------------------------------------------------------------------------

typedef __bf16 bf16;
typedef __attribute__((ext_vector_type(16))) __bf16 v16bf;
typedef __attribute__((ext_vector_type(8)))  __bf16 v8bf;
typedef __attribute__((ext_vector_type(8)))  float  v8f;

#define NUM_LAYERS 10
#define HID   512
#define BATCH 128
#define SEQ   64
#define GATES 2048   // 4*HID
#define KDIM  1024   // HID (x/cur) + HID (h)

// LDS layout (bytes): A tile [16 rows][1040 elems] (padded: 2080B/row), then
// B slabs [4 gates][32 kb][32 lanes][16 elems] (32768B per gate).
#define LDS_A_ROW   2080
#define LDS_B_OFF   (16 * LDS_A_ROW)          // 33280
#define LDS_TOTAL   (LDS_B_OFF + 4 * 32768)   // 164352 (< 320KB)

__device__ __forceinline__ unsigned short f2bf(float f) {
  unsigned int u = __float_as_uint(f);
  u += 0x7FFFu + ((u >> 16) & 1u);   // round-to-nearest-even
  return (unsigned short)(u >> 16);
}
__device__ __forceinline__ float fsig(float x)  { return 1.0f / (1.0f + __expf(-x)); }
__device__ __forceinline__ float ftanh(float x) { return 2.0f * fsig(2.0f * x) - 1.0f; }

// Async DMA: 16 bytes global -> LDS, tracked by ASYNCcnt (no data VGPRs, so
// the scheduler cannot serialize it against WMMA consumers).
__device__ __forceinline__ void async_copy16(unsigned lds_byte, const void* gptr) {
  asm volatile("global_load_async_to_lds_b128 %0, %1, off"
               :: "v"(lds_byte), "v"(gptr) : "memory");
}
__device__ __forceinline__ void wait_async0() {
  asm volatile("s_wait_asynccnt 0x0" ::: "memory");
}

// ---------------------------------------------------------------------------
// Pack [w_ih ; w_hh] (fp32, row-major [2048][512] each) into bf16 WMMA
// B-operand layout: Wp[l][g][nb][kb][lane][n] = W[g*512+nb*16+lane%16][kb*32+16*(n/8)+8*(lane/16)+n%8]
// => per (g,nb): one contiguous 32KB slab; per lane one contiguous 32B frag.
// ---------------------------------------------------------------------------
__global__ __launch_bounds__(256) void pack_weights_kernel(
    const float* __restrict__ w_ih, const float* __restrict__ w_hh,
    unsigned short* __restrict__ Wp) {
  size_t idx = (size_t)blockIdx.x * 256 + threadIdx.x;
  const size_t total = (size_t)NUM_LAYERS * 4 * 32 * 32 * 32 * 16;
  if (idx >= total) return;
  int n    = (int)(idx & 15);  size_t r = idx >> 4;
  int lane = (int)(r & 31);    r >>= 5;
  int kb   = (int)(r & 31);    r >>= 5;
  int nb   = (int)(r & 31);    r >>= 5;
  int g    = (int)(r & 3);     r >>= 2;
  int l    = (int)r;
  int row = g * HID + nb * 16 + (lane & 15);   // gate row 0..2047
  int k   = kb * 32 + ((n & 8) << 1) + ((lane >> 4) << 3) + (n & 7);          // 0..1023
  float v = (k < HID)
      ? w_ih[((size_t)l * GATES + row) * HID + k]
      : w_hh[((size_t)l * GATES + row) * HID + (k - HID)];
  Wp[idx] = f2bf(v);
}

// Combine biases, zero c-state, zero h-halves of the parity-0 activation bufs.
__global__ __launch_bounds__(256) void init_state_kernel(
    const float* __restrict__ b_ih, const float* __restrict__ b_hh,
    float* __restrict__ bias, float* __restrict__ c_state,
    unsigned short* __restrict__ Abuf) {
  int idx = blockIdx.x * 256 + threadIdx.x;   // 0 .. 655359
  if (idx < NUM_LAYERS * GATES) bias[idx] = b_ih[idx] + b_hh[idx];
  if (idx < NUM_LAYERS * BATCH * HID) {
    c_state[idx] = 0.0f;
    int l   = idx / (BATCH * HID);
    int rm  = idx - l * (BATCH * HID);
    int b   = rm >> 9;
    int col = rm & 511;
    Abuf[(size_t)(l * 2) * BATCH * KDIM + (size_t)b * KDIM + HID + col] = 0;
  }
}

// ---------------------------------------------------------------------------
// Fused GEMM (bf16 WMMA, 8-way split-K out of LDS) + LSTM gate pointwise.
// Block = 256 threads (8 waves), tile = 16 batch x 16 hidden (x 4 gates).
// Grid = (32 hidden tiles, 8 batch tiles) = 256 blocks, 2048 waves.
// ---------------------------------------------------------------------------
__global__ __launch_bounds__(256) void lstm_layer_kernel(
    const unsigned short* __restrict__ Ard,   // [128][1024] bf16 (this parity)
    unsigned short* __restrict__ Ahwr,        // other-parity buf: h-half dest
    unsigned short* __restrict__ Anext,       // next layer's input half (or null)
    const float* __restrict__ x0,             // layer 0 only: x_t fp32, else null
    const unsigned short* __restrict__ Wp,    // layer weights [4][32][32][32][16]
    const float* __restrict__ bias,           // [2048] (b_ih+b_hh)
    float* __restrict__ c_state,              // [128][512]
    float* __restrict__ outp) {               // d_out on final step, else null
  __shared__ __align__(32) char smem[LDS_TOTAL];
  const int tid  = threadIdx.x;
  const int wave = tid >> 5;
  const int lane = tid & 31;
  const int nb = blockIdx.x;   // hidden tile 0..31
  const int bm = blockIdx.y;   // batch tile  0..7

  // ---- Phase 1: stage the block's working set into LDS. ------------------
  const unsigned sbase = (unsigned)(size_t)smem;   // wave-relative LDS byte addr
  // B slabs: per gate one contiguous 32KB block at (g*32+nb)*32768 bytes.
  #pragma unroll
  for (int g = 0; g < 4; ++g) {
    const char* Bsrc = (const char*)Wp + ((size_t)(g * 32 + nb) << 15);
    const unsigned dst0 = sbase + LDS_B_OFF + (unsigned)g * 32768u;
    #pragma unroll
    for (int i = 0; i < 8; ++i) {
      const unsigned off = (unsigned)(tid + i * 256) * 16;
      async_copy16(dst0 + off, Bsrc + off);
    }
  }
  if (x0) {
    // Layer 0: input half = x_t tile (contiguous 32KB fp32) converted inline.
    const char* Xsrc = (const char*)(x0 + (size_t)bm * 16 * HID);
    #pragma unroll
    for (int i = 0; i < 8; ++i) {
      const unsigned off = (unsigned)(tid + i * 256) * 16;  // 0..32752
      const unsigned row = off >> 11;                       // fp32 row = 2048B
      const unsigned within = off & 2047;
      const float4 f = *(const float4*)(Xsrc + off);
      const unsigned lo = (unsigned)f2bf(f.x) | ((unsigned)f2bf(f.y) << 16);
      const unsigned hi = (unsigned)f2bf(f.z) | ((unsigned)f2bf(f.w) << 16);
      *(uint2*)(smem + row * LDS_A_ROW + (within >> 1)) = make_uint2(lo, hi);
    }
    // h half: 16 chunks of 1KB (strided per row) via async copy.
    #pragma unroll
    for (int i = 0; i < 4; ++i) {
      const unsigned idx = (unsigned)(tid + i * 256);       // 0..1023
      const unsigned row = idx >> 6;                        // 0..15
      const unsigned within = (idx & 63) * 16;              // 0..1008
      const char* src =
          (const char*)(Ard + (size_t)(bm * 16 + row) * KDIM + HID) + within;
      async_copy16(sbase + row * LDS_A_ROW + 1024 + within, src);
    }
  } else {
    // A tile: 16 consecutive rows of Ard => one contiguous 32KB global block.
    const char* Asrc = (const char*)(Ard + (size_t)bm * 16 * KDIM);
    #pragma unroll
    for (int i = 0; i < 8; ++i) {
      const unsigned off = (unsigned)(tid + i * 256) * 16;   // 0..32752
      const unsigned row = off >> 11;                        // /2048
      const unsigned within = off & 2047;
      async_copy16(sbase + row * LDS_A_ROW + within, Asrc + off);
    }
  }
  wait_async0();        // this wave's async transfers done
  __syncthreads();      // everyone's transfers (and ds stores) visible

  // ---- Phase 2: 16 WMMAs per wave, operands from LDS. -------------------
  v8f acc0 = {}, acc1 = {}, acc2 = {}, acc3 = {};
  // A fragment (16x32 bf16): lane holds row m=lane%16;
  // lanes 0-15 hold K {0-7,16-23}, lanes 16-31 hold K {8-15,24-31}.
  const int m   = lane & 15;
  const int sel = (lane >> 4) << 3;   // 0 or 8
  const bf16* lA = (const bf16*)smem;                 // [16][1040] padded rows
  const bf16* lB = (const bf16*)(smem + LDS_B_OFF);   // [4][32][32][16]
  const bf16* lArow = lA + (size_t)m * (LDS_A_ROW / 2);
  const bf16* lBlane = lB + (size_t)lane * 16;

  const int kb0 = wave * 4;            // split-K: wave owns one K-eighth
  #pragma unroll
  for (int kk = 0; kk < 4; ++kk) {
    const int kb = kb0 + kk;
    const int kbase = kb * 32;
    union { v16bf v; v8bf h[2]; } au;
    au.h[0] = *(const v8bf*)(lArow + kbase + sel);
    au.h[1] = *(const v8bf*)(lArow + kbase + 16 + sel);
    const v16bf a = au.v;
    const bf16* wk = lBlane + (size_t)kb * 512;
    const v16bf b0 = *(const v16bf*)(wk);
    const v16bf b1 = *(const v16bf*)(wk + 16384);
    const v16bf b2 = *(const v16bf*)(wk + 2 * 16384);
    const v16bf b3 = *(const v16bf*)(wk + 3 * 16384);
    acc0 = __builtin_amdgcn_wmma_f32_16x16x32_bf16(false, a, false, b0, (short)0, acc0, false, false);
    acc1 = __builtin_amdgcn_wmma_f32_16x16x32_bf16(false, a, false, b1, (short)0, acc1, false, false);
    acc2 = __builtin_amdgcn_wmma_f32_16x16x32_bf16(false, a, false, b2, (short)0, acc2, false, false);
    acc3 = __builtin_amdgcn_wmma_f32_16x16x32_bf16(false, a, false, b3, (short)0, acc3, false, false);
  }

  // ---- Phase 3: split-K reduce + LSTM pointwise. Reduction buffer aliases
  // the (now dead) staging LDS; the barrier below separates the lifetimes. --
  __syncthreads();
  float* red = (float*)smem;          // [8 waves][4 gates][8 vreg][32 lane]
  #pragma unroll
  for (int v = 0; v < 8; ++v) {
    red[((wave * 4 + 0) * 8 + v) * 32 + lane] = acc0[v];
    red[((wave * 4 + 1) * 8 + v) * 32 + lane] = acc1[v];
    red[((wave * 4 + 2) * 8 + v) * 32 + lane] = acc2[v];
    red[((wave * 4 + 3) * 8 + v) * 32 + lane] = acc3[v];
  }
  __syncthreads();

  {  // 256 elements over 256 threads; C/D elem (v,l): m=v+8*(l/16), n=l%16.
    const int e  = tid;
    const int v  = e >> 5;
    const int l2 = e & 31;
    float gi = 0.f, gf = 0.f, gg = 0.f, go = 0.f;
    #pragma unroll
    for (int w = 0; w < 8; ++w) {
      gi += red[((w * 4 + 0) * 8 + v) * 32 + l2];
      gf += red[((w * 4 + 1) * 8 + v) * 32 + l2];
      gg += red[((w * 4 + 2) * 8 + v) * 32 + l2];
      go += red[((w * 4 + 3) * 8 + v) * 32 + l2];
    }
    const int mm  = v + ((l2 >> 4) << 3);
    const int n   = l2 & 15;
    const int b   = bm * 16 + mm;
    const int col = nb * 16 + n;
    gi += bias[col];
    gf += bias[HID + col];
    gg += bias[2 * HID + col];
    go += bias[3 * HID + col];
    const float cold = c_state[b * HID + col];
    const float cn = fsig(gf) * cold + fsig(gi) * ftanh(gg);
    const float hn = fsig(go) * ftanh(cn);
    c_state[b * HID + col] = cn;
    Ahwr[(size_t)b * KDIM + HID + col] = f2bf(hn);        // h -> next timestep
    if (Anext) Anext[(size_t)b * KDIM + col] = f2bf(cn);  // cur = c (ref quirk)
    if (outp)  outp[b * HID + col] = cn;                  // cs[-1] final output
  }
}

// ---------------------------------------------------------------------------
extern "C" void kernel_launch(void* const* d_in, const int* in_sizes, int n_in,
                              void* d_out, int out_size, void* d_ws, size_t ws_size,
                              hipStream_t stream) {
  const float* inputs = (const float*)d_in[0];
  const float* w_ih   = (const float*)d_in[1];
  const float* w_hh   = (const float*)d_in[2];
  const float* b_ih   = (const float*)d_in[3];
  const float* b_hh   = (const float*)d_in[4];
  float* out = (float*)d_out;

  // Workspace carve-up (~47.6 MB total).
  char* ws = (char*)d_ws;
  const size_t WP_ELEMS = (size_t)NUM_LAYERS * 2097152;       // 4*32*32*32*16 per layer
  unsigned short* Wp = (unsigned short*)ws;
  size_t off = WP_ELEMS * 2;                                  // 41,943,040 B
  float* bias = (float*)(ws + off);
  off += (size_t)NUM_LAYERS * GATES * 4;                      // +81,920 B
  float* c_state = (float*)(ws + off);
  off += (size_t)NUM_LAYERS * BATCH * HID * 4;                // +2,621,440 B
  unsigned short* Abuf = (unsigned short*)(ws + off);         // +5,242,880 B

  pack_weights_kernel<<<(unsigned)(WP_ELEMS / 256), 256, 0, stream>>>(w_ih, w_hh, Wp);
  init_state_kernel<<<(NUM_LAYERS * BATCH * HID) / 256, 256, 0, stream>>>(
      b_ih, b_hh, bias, c_state, Abuf);

  const size_t ABUF_PAR   = (size_t)BATCH * KDIM;   // one parity buffer
  const size_t ABUF_LAYER = 2 * ABUF_PAR;           // per layer

  for (int t = 0; t < SEQ; ++t) {
    const int p = t & 1;
    for (int j = 0; j < NUM_LAYERS; ++j) {
      unsigned short* Ard   = Abuf + (size_t)j * ABUF_LAYER + (size_t)p * ABUF_PAR;
      unsigned short* Ahwr  = Abuf + (size_t)j * ABUF_LAYER + (size_t)(1 - p) * ABUF_PAR;
      unsigned short* Anext = (j + 1 < NUM_LAYERS)
          ? Abuf + (size_t)(j + 1) * ABUF_LAYER + (size_t)p * ABUF_PAR : nullptr;
      const float* x0 = (j == 0) ? inputs + (size_t)t * BATCH * HID : nullptr;
      const unsigned short* Wpl = Wp + (size_t)j * 2097152;
      float* cs = c_state + (size_t)j * BATCH * HID;
      float* op = (j == NUM_LAYERS - 1 && t == SEQ - 1) ? out : nullptr;
      lstm_layer_kernel<<<dim3(32, 8), 256, 0, stream>>>(
          Ard, Ahwr, Anext, x0, Wpl, bias + j * GATES, cs, op);
    }
  }
}
